// MultiHeadAttention_88158498717899
// MI455X (gfx1250) — matrix operand order repaired and checked by reading, not compiled, not run on hardware
//
#include <hip/hip_runtime.h>
#include <hip/hip_bf16.h>
#include <math.h>

// ---------------------------------------------------------------------------
// MI455X (gfx1250) fused MHA: wave32, V_WMMA_F32_16X16X32_F16 compute,
// Tensor Data Mover (TDM) K/V staging, async global->LDS GEMM staging.
// ---------------------------------------------------------------------------

typedef _Float16 f16;
typedef __attribute__((ext_vector_type(16))) _Float16 v16h;
typedef __attribute__((ext_vector_type(8)))  float    v8f;
typedef __attribute__((ext_vector_type(4)))  unsigned int u32x4;
typedef __attribute__((ext_vector_type(8)))  int i32x8;
typedef __attribute__((ext_vector_type(4)))  int i32x4;

#define NB 4
#define NL 2048
#define NE 1024
#define NH 8
#define ND 128
#define NM (NB * NL)          // 8192 rows
#define ATT_SCALE 0.08838834764831845f   // 1/sqrt(128)

union Frag { unsigned int u[8]; v16h v; };
union Acc  { float f[8]; v8f v; };

// K-index (within a 32-wide chunk) of A-fragment uint pair v for lane-group g.
__device__ __forceinline__ int akk(int v, int g) {
    return (v < 4 ? v * 2 : 16 + (v - 4) * 2) + g * 8;
}

// Per-lane async global->LDS 16-byte copy (ASYNCcnt).  lds = generic pointer
// into __shared__ (low 32 bits of the flat address are the LDS byte address).
#define ASYNC_B128(ldsPtr, basePtr, byteOff)                                   \
    asm volatile("global_load_async_to_lds_b128 %0, %1, %2"                    \
                 :: "v"((unsigned)(unsigned long long)(ldsPtr)),               \
                    "v"((int)(byteOff)), "s"(basePtr) : "memory")

#define WAIT_ASYNC() asm volatile("s_wait_asynccnt 0x0" ::: "memory")

// ---------------------------------------------------------------------------
// TDM 2D tile load: tensor of `tensor_d1` lines x `tensor_d0` f16 elements
// (line stride `stride0` elements); copies tile_d1 x tile_d0 from `gptr`
// (tile start) into LDS at `lds_byte`, packed contiguously.
// D# packing per CDNA5 ISA ch.8 (groups 0 and 1; groups 2/3 zero => 2D).
// ---------------------------------------------------------------------------
__device__ __forceinline__ void tdm_load_2d(
    const void* gptr, unsigned lds_byte,
    unsigned tensor_d0, unsigned tensor_d1,
    unsigned tile_d0, unsigned tile_d1, unsigned stride0)
{
    unsigned long long ga = (unsigned long long)gptr;
    u32x4 g0;
    g0.x = 1u;                                        // count=1 (valid D#)
    g0.y = lds_byte;                                  // lds_addr [63:32]
    g0.z = (unsigned)(ga & 0xffffffffull);            // global_addr lo
    g0.w = (unsigned)((ga >> 32) & 0x1ffffffull)      // global_addr hi (57b)
         | (2u << 30);                                // type = 2 ("image")
    i32x8 g1;
    g1[0] = (int)(1u << 16);                          // data_size=1 -> 2 bytes
    g1[1] = (int)((tensor_d0 & 0xffffu) << 16);       // tensor_dim0 lo16
    g1[2] = (int)(((tensor_d0 >> 16) & 0xffffu)       // tensor_dim0 hi16
         | ((tensor_d1 & 0xffffu) << 16));            // tensor_dim1 lo16
    g1[3] = (int)(((tensor_d1 >> 16) & 0xffffu)       // tensor_dim1 hi16
         | ((tile_d0 & 0xffffu) << 16));              // tile_dim0
    g1[4] = (int)(tile_d1 & 0xffffu);                 // tile_dim1 (tile_dim2=0)
    g1[5] = (int)stride0;                             // tensor_dim0_stride lo32
    g1[6] = 0;                                        // stride0 hi16 | stride1 lo16
    g1[7] = 0;                                        // stride1 hi32
    i32x4 gz4 = {};
    i32x8 gz8 = {};
    __builtin_amdgcn_tensor_load_to_lds(g0, g1, gz4, gz4, gz8, 0);
}

// ---------------------------------------------------------------------------
// Kernel 0: fp32 weight [K=1024][N=1024]  ->  f16 transposed [N][K]
// ---------------------------------------------------------------------------
__global__ __launch_bounds__(256) void wtrans_kernel(
    const float* __restrict__ w0, const float* __restrict__ w1,
    const float* __restrict__ w2, const float* __restrict__ w3,
    f16* __restrict__ o0, f16* __restrict__ o1,
    f16* __restrict__ o2, f16* __restrict__ o3)
{
    __shared__ f16 tile[32][33];
    const float* src; f16* dst;
    switch (blockIdx.z) {
        case 0:  src = w0; dst = o0; break;
        case 1:  src = w1; dst = o1; break;
        case 2:  src = w2; dst = o2; break;
        default: src = w3; dst = o3; break;
    }
    int n0 = blockIdx.x * 32, k0 = blockIdx.y * 32;
    int tx = threadIdx.x, ty = threadIdx.y;          // (32, 8)
    #pragma unroll
    for (int i = 0; i < 4; ++i)
        tile[ty + i * 8][tx] = (f16)src[(size_t)(k0 + ty + i * 8) * 1024 + n0 + tx];
    __syncthreads();
    #pragma unroll
    for (int i = 0; i < 4; ++i)
        dst[(size_t)(n0 + ty + i * 8) * 1024 + k0 + tx] = tile[tx][ty + i * 8];
}

// ---------------------------------------------------------------------------
// Kernel 1: QKV projection GEMM.  z=0: Q -> [b][h][l][d]
//                                 z=1: K -> [b][h][l][d]
//                                 z=2: V -> [b][h][d][l] (transposed)
// A tile: fp32->f16 convert through VGPRs; B tile: async global->LDS.
// ---------------------------------------------------------------------------
__global__ __launch_bounds__(256) void qkv_kernel(
    const float* __restrict__ q_in, const float* __restrict__ kv_in,
    const f16* __restrict__ wt_q, const f16* __restrict__ wt_k,
    const f16* __restrict__ wt_v,
    f16* __restrict__ q16, f16* __restrict__ k16, f16* __restrict__ vt16)
{
    __shared__ __align__(16) f16 As[128 * 32];   // [m][k]
    __shared__ __align__(16) f16 Bt[128 * 32];   // [n][k]

    const int z  = blockIdx.z;
    const float* A  = (z == 0) ? q_in : kv_in;
    const f16*   Wt = (z == 0) ? wt_q : (z == 1 ? wt_k : wt_v);
    const int n0 = blockIdx.x * 128;
    const int m0 = blockIdx.y * 128;
    const int tid = threadIdx.x;
    const int lane = tid & 31, w = tid >> 5;
    const int wm = w >> 1, wn = w & 1;
    const int hl = lane & 15, g = lane >> 4;

    Acc acc[2][4];
    v8f zero = 0;
    #pragma unroll
    for (int mt = 0; mt < 2; ++mt)
        #pragma unroll
        for (int nt = 0; nt < 4; ++nt) acc[mt][nt].v = zero;

    for (int k0 = 0; k0 < 1024; k0 += 32) {
        // B tile: 512 x uint4, async DMA to LDS.
        #pragma unroll
        for (int i = 0; i < 2; ++i) {
            int s = tid + i * 256;
            int n = s >> 2, kq = s & 3;
            ASYNC_B128(&Bt[n * 32 + kq * 8], Wt,
                       ((size_t)(n0 + n) * 1024 + k0 + kq * 8) * 2);
        }
        // A tile: 128x32 fp32 -> f16 (1024 float4 slots).
        #pragma unroll
        for (int i = 0; i < 4; ++i) {
            int s = tid + i * 256;
            int r = s >> 3, c4 = s & 7;
            const float4 f = *(const float4*)&A[(size_t)(m0 + r) * 1024 + k0 + c4 * 4];
            union { f16 h[4]; unsigned long long u; } pk;
            pk.h[0] = (f16)f.x; pk.h[1] = (f16)f.y;
            pk.h[2] = (f16)f.z; pk.h[3] = (f16)f.w;
            *(unsigned long long*)&As[r * 32 + c4 * 4] = pk.u;
        }
        WAIT_ASYNC();
        __syncthreads();

        Frag af[2];
        #pragma unroll
        for (int mt = 0; mt < 2; ++mt) {
            int r = wm * 32 + mt * 16 + hl;
            #pragma unroll
            for (int v = 0; v < 8; ++v)
                af[mt].u[v] = *(const unsigned int*)&As[r * 32 + akk(v, g)];
        }
        #pragma unroll
        for (int nt = 0; nt < 4; ++nt) {
            Frag bf;
            int n = wn * 64 + nt * 16 + hl;
            #pragma unroll
            for (int v = 0; v < 8; ++v)
                bf.u[v] = *(const unsigned int*)&Bt[n * 32 + akk(v, g)];
            #pragma unroll
            for (int mt = 0; mt < 2; ++mt)
                acc[mt][nt].v = __builtin_amdgcn_wmma_f32_16x16x32_f16(
                    false, af[mt].v, false, bf.v, (short)0, acc[mt][nt].v,
                    false, false);
        }
        __syncthreads();
    }

    // Epilogue: z is block-uniform -> one branch, no per-store control flow.
    if (z == 2) {
        #pragma unroll
        for (int mt = 0; mt < 2; ++mt)
            #pragma unroll
            for (int nt = 0; nt < 4; ++nt)
                #pragma unroll
                for (int v = 0; v < 8; ++v) {
                    int row = m0 + wm * 32 + mt * 16 + v + 8 * g;
                    int col = n0 + wn * 64 + nt * 16 + hl;
                    int b = row >> 11, l = row & 2047;
                    int hh = col >> 7, d = col & 127;
                    vt16[((size_t)(b * NH + hh) * ND + d) * NL + l] =
                        (f16)acc[mt][nt].f[v];
                }
    } else {
        f16* o = (z == 0) ? q16 : k16;
        #pragma unroll
        for (int mt = 0; mt < 2; ++mt)
            #pragma unroll
            for (int nt = 0; nt < 4; ++nt)
                #pragma unroll
                for (int v = 0; v < 8; ++v) {
                    int row = m0 + wm * 32 + mt * 16 + v + 8 * g;
                    int col = n0 + wn * 64 + nt * 16 + hl;
                    int b = row >> 11, l = row & 2047;
                    int hh = col >> 7, d = col & 127;
                    o[((size_t)(b * NH + hh) * NL + l) * ND + d] =
                        (f16)acc[mt][nt].f[v];
                }
    }
}

// ---------------------------------------------------------------------------
// Kernel 2: flash attention.  WG = 128 q-rows of one (b,h); 8 waves x 16 rows.
// K/V tiles DMA'd by the Tensor Data Mover, double-buffered: the TDM for key
// block i+1 runs under the 32 WMMAs + softmax of block i (TENSORcnt overlap).
// ---------------------------------------------------------------------------
__global__ __launch_bounds__(256) void attn_kernel(
    const f16* __restrict__ q16, const f16* __restrict__ k16,
    const f16* __restrict__ vt16, const unsigned char* __restrict__ mask,
    f16* __restrict__ ctx16)
{
    __shared__ __align__(16) f16 Kl[2][64 * 128];   // [key][d]     2 x 16KB
    __shared__ __align__(16) f16 Vl[2][128 * 64];   // [d][key]     2 x 16KB
    __shared__ __align__(16) f16 Pb[8][16 * 64];    // per-wave P   16KB

    const int tid = threadIdx.x, lane = tid & 31, w = tid >> 5;
    const int hl = lane & 15, g = lane >> 4;
    const int bh = blockIdx.y;                 // b*8 + h
    const int b = bh >> 3, h = bh & 7;
    const int q0 = blockIdx.x * 128;
    const f16* Qh = q16 + (size_t)bh * NL * ND;
    const f16* Kh = k16 + (size_t)bh * NL * ND;
    const f16* Vh = vt16 + (size_t)bh * ND * NL;
    const unsigned char* Mh = mask + (size_t)bh * NL * NL;

    // Q fragments: loaded once, reused for every key block.
    const int qrow = q0 + w * 16 + hl;
    Frag qf[4];
    #pragma unroll
    for (int c = 0; c < 4; ++c)
        #pragma unroll
        for (int v = 0; v < 8; ++v)
            qf[c].u[v] = *(const unsigned int*)&Qh[(size_t)qrow * ND + c * 32 + akk(v, g)];

    float m[8], lsum[8];
    Acc oacc[8];
    v8f zero = 0;
    #pragma unroll
    for (int v = 0; v < 8; ++v) { m[v] = -1e30f; lsum[v] = 0.f; }
    #pragma unroll
    for (int nt = 0; nt < 8; ++nt) oacc[nt].v = zero;

    // TDM issue for one key block (wave-0 only; EXEC-independent DMA).
    // K:  tensor [2048 keys][128 d], tile 64x128 -> Kl[buf]
    // Vt: tensor [128 d][2048 keys], tile 128x64 -> Vl[buf]
    auto issue_kv = [&](int kb, int buf) {
        tdm_load_2d(Kh + (size_t)kb * ND,
                    (unsigned)(unsigned long long)&Kl[buf][0],
                    /*tensor_d0=*/ND, /*tensor_d1=*/NL,
                    /*tile_d0=*/ND, /*tile_d1=*/64, /*stride0=*/ND);
        tdm_load_2d(Vh + kb,
                    (unsigned)(unsigned long long)&Vl[buf][0],
                    /*tensor_d0=*/NL, /*tensor_d1=*/ND,
                    /*tile_d0=*/64, /*tile_d1=*/ND, /*stride0=*/NL);
    };

    if (w == 0) issue_kv(0, 0);

    for (int kb = 0; kb < NL; kb += 64) {
        const int cur = (kb >> 6) & 1;
        if (w == 0) __builtin_amdgcn_s_wait_tensorcnt(0);
        __syncthreads();
        if (w == 0 && kb + 64 < NL) issue_kv(kb + 64, cur ^ 1);

        const f16* Kc = &Kl[cur][0];
        const f16* Vc = &Vl[cur][0];

        // S = Q K^T : 4 d-chunks x 4 key-tiles.
        Acc sc[4];
        #pragma unroll
        for (int t = 0; t < 4; ++t) sc[t].v = zero;
        #pragma unroll
        for (int c = 0; c < 4; ++c)
            #pragma unroll
            for (int t = 0; t < 4; ++t) {
                Frag kf;
                int key = t * 16 + hl;
                #pragma unroll
                for (int v = 0; v < 8; ++v)
                    kf.u[v] = *(const unsigned int*)&Kc[key * 128 + c * 32 + akk(v, g)];
                sc[t].v = __builtin_amdgcn_wmma_f32_16x16x32_f16(
                    false, qf[c].v, false, kf.v, (short)0, sc[t].v, false, false);
            }

        // Mask fill (-1e9, as in reference) + scale.
        #pragma unroll
        for (int t = 0; t < 4; ++t)
            #pragma unroll
            for (int v = 0; v < 8; ++v) {
                int qq = q0 + w * 16 + v + 8 * g;
                int kk = kb + t * 16 + hl;
                unsigned char mb = Mh[(size_t)qq * NL + kk];
                sc[t].f[v] = mb ? -1e9f : sc[t].f[v] * ATT_SCALE;
            }

        // Online softmax (rows live across 16 lanes).
        #pragma unroll
        for (int v = 0; v < 8; ++v) {
            float r = fmaxf(fmaxf(sc[0].f[v], sc[1].f[v]),
                            fmaxf(sc[2].f[v], sc[3].f[v]));
            r = fmaxf(r, __shfl_xor(r, 1, 16));
            r = fmaxf(r, __shfl_xor(r, 2, 16));
            r = fmaxf(r, __shfl_xor(r, 4, 16));
            r = fmaxf(r, __shfl_xor(r, 8, 16));
            float mn = fmaxf(m[v], r);
            float alpha = __expf(m[v] - mn);
            m[v] = mn;
            lsum[v] *= alpha;
            #pragma unroll
            for (int nt = 0; nt < 8; ++nt) oacc[nt].f[v] *= alpha;
        }

        // P = exp(S - m): register row-sums, f16 staging to per-wave LDS.
        #pragma unroll
        for (int v = 0; v < 8; ++v) {
            float rs = 0.f;
            #pragma unroll
            for (int t = 0; t < 4; ++t) {
                float p = __expf(sc[t].f[v] - m[v]);
                rs += p;
                Pb[w][(v + 8 * g) * 64 + t * 16 + hl] = (f16)p;
            }
            rs += __shfl_xor(rs, 1, 16);
            rs += __shfl_xor(rs, 2, 16);
            rs += __shfl_xor(rs, 4, 16);
            rs += __shfl_xor(rs, 8, 16);
            lsum[v] += rs;
        }

        // O += P V : 2 key-chunks x 8 dv-tiles.
        #pragma unroll
        for (int c = 0; c < 2; ++c) {
            Frag pf;
            #pragma unroll
            for (int v = 0; v < 8; ++v)
                pf.u[v] = *(const unsigned int*)&Pb[w][hl * 64 + c * 32 + akk(v, g)];
            #pragma unroll
            for (int nt = 0; nt < 8; ++nt) {
                Frag vf;
                #pragma unroll
                for (int v = 0; v < 8; ++v)
                    vf.u[v] = *(const unsigned int*)
                        &Vc[(nt * 16 + hl) * 64 + c * 32 + akk(v, g)];
                oacc[nt].v = __builtin_amdgcn_wmma_f32_16x16x32_f16(
                    false, pf.v, false, vf.v, (short)0, oacc[nt].v, false, false);
            }
        }
    }

    // Normalize, emit ctx f16 [B][L][H*DV].
    #pragma unroll
    for (int v = 0; v < 8; ++v) {
        float inv = 1.0f / lsum[v];
        int qq = q0 + w * 16 + v + 8 * g;
        #pragma unroll
        for (int nt = 0; nt < 8; ++nt)
            ctx16[((size_t)b * NL + qq) * (NH * ND) + h * ND + nt * 16 + hl] =
                (f16)(oacc[nt].f[v] * inv);
    }
}

// ---------------------------------------------------------------------------
// Kernel 3: output projection  out = ctx16 @ Wfc  (f16 x f16 -> fp32).
// Double-buffered async global->LDS tile staging (ASYNCcnt overlap).
// ---------------------------------------------------------------------------
__global__ __launch_bounds__(256) void proj_kernel(
    const f16* __restrict__ ctx16, const f16* __restrict__ wt_fc,
    float* __restrict__ out)
{
    __shared__ __align__(16) f16 As[2][128 * 32];
    __shared__ __align__(16) f16 Bt[2][128 * 32];

    const int n0 = blockIdx.x * 128;
    const int m0 = blockIdx.y * 128;
    const int tid = threadIdx.x;
    const int lane = tid & 31, w = tid >> 5;
    const int wm = w >> 1, wn = w & 1;
    const int hl = lane & 15, g = lane >> 4;

    Acc acc[2][4];
    v8f zero = 0;
    #pragma unroll
    for (int mt = 0; mt < 2; ++mt)
        #pragma unroll
        for (int nt = 0; nt < 4; ++nt) acc[mt][nt].v = zero;

    auto issue = [&](int k0, int buf) {
        #pragma unroll
        for (int i = 0; i < 2; ++i) {
            int s = tid + i * 256;
            int r = s >> 2, kq = s & 3;
            ASYNC_B128(&As[buf][r * 32 + kq * 8], ctx16,
                       ((size_t)(m0 + r) * 1024 + k0 + kq * 8) * 2);
        }
        #pragma unroll
        for (int i = 0; i < 2; ++i) {
            int s = tid + i * 256;
            int n = s >> 2, kq = s & 3;
            ASYNC_B128(&Bt[buf][n * 32 + kq * 8], wt_fc,
                       ((size_t)(n0 + n) * 1024 + k0 + kq * 8) * 2);
        }
    };

    issue(0, 0);

    for (int k0 = 0; k0 < 1024; k0 += 32) {
        const int cur = (k0 >> 5) & 1;
        WAIT_ASYNC();
        __syncthreads();
        if (k0 + 32 < 1024) issue(k0 + 32, cur ^ 1);

        Frag af[2];
        #pragma unroll
        for (int mt = 0; mt < 2; ++mt) {
            int r = wm * 32 + mt * 16 + hl;
            #pragma unroll
            for (int v = 0; v < 8; ++v)
                af[mt].u[v] = *(const unsigned int*)&As[cur][r * 32 + akk(v, g)];
        }
        #pragma unroll
        for (int nt = 0; nt < 4; ++nt) {
            Frag bf;
            int n = wn * 64 + nt * 16 + hl;
            #pragma unroll
            for (int v = 0; v < 8; ++v)
                bf.u[v] = *(const unsigned int*)&Bt[cur][n * 32 + akk(v, g)];
            #pragma unroll
            for (int mt = 0; mt < 2; ++mt)
                acc[mt][nt].v = __builtin_amdgcn_wmma_f32_16x16x32_f16(
                    false, af[mt].v, false, bf.v, (short)0, acc[mt][nt].v,
                    false, false);
        }
    }

    #pragma unroll
    for (int mt = 0; mt < 2; ++mt)
        #pragma unroll
        for (int nt = 0; nt < 4; ++nt)
            #pragma unroll
            for (int v = 0; v < 8; ++v) {
                int row = m0 + wm * 32 + mt * 16 + v + 8 * g;
                int col = n0 + wn * 64 + nt * 16 + hl;
                out[(size_t)row * 1024 + col] = acc[mt][nt].f[v];
            }
}

// ---------------------------------------------------------------------------
// Host launcher
// ---------------------------------------------------------------------------
extern "C" void kernel_launch(void* const* d_in, const int* in_sizes, int n_in,
                              void* d_out, int out_size, void* d_ws, size_t ws_size,
                              hipStream_t stream)
{
    (void)in_sizes; (void)n_in; (void)out_size; (void)ws_size;
    const float* q_in  = (const float*)d_in[0];
    const float* kv_in = (const float*)d_in[1];
    const unsigned char* mask = (const unsigned char*)d_in[2];
    const float* w_q  = (const float*)d_in[3];
    const float* w_k  = (const float*)d_in[4];
    const float* w_v  = (const float*)d_in[5];
    const float* w_fc = (const float*)d_in[6];

    char* ws = (char*)d_ws;                       // 72 MB used
    f16* wt_q  = (f16*)(ws + (size_t) 0 * (1 << 20));
    f16* wt_k  = (f16*)(ws + (size_t) 2 * (1 << 20));
    f16* wt_v  = (f16*)(ws + (size_t) 4 * (1 << 20));
    f16* wt_fc = (f16*)(ws + (size_t) 6 * (1 << 20));
    f16* q16   = (f16*)(ws + (size_t) 8 * (1 << 20));
    f16* k16   = (f16*)(ws + (size_t)24 * (1 << 20));
    f16* vt16  = (f16*)(ws + (size_t)40 * (1 << 20));
    f16* c16   = (f16*)(ws + (size_t)56 * (1 << 20));

    hipLaunchKernelGGL(wtrans_kernel, dim3(32, 32, 4), dim3(32, 8, 1), 0, stream,
                       w_q, w_k, w_v, w_fc, wt_q, wt_k, wt_v, wt_fc);
    hipLaunchKernelGGL(qkv_kernel, dim3(8, 64, 3), dim3(256, 1, 1), 0, stream,
                       q_in, kv_in, wt_q, wt_k, wt_v, q16, k16, vt16);
    hipLaunchKernelGGL(attn_kernel, dim3(16, 32, 1), dim3(256, 1, 1), 0, stream,
                       q16, k16, vt16, mask, c16);
    hipLaunchKernelGGL(proj_kernel, dim3(8, 64, 1), dim3(256, 1, 1), 0, stream,
                       c16, wt_fc, (float*)d_out);
}
